// BiLSTMCRF_65481071394986
// MI455X (gfx1250) — matrix operand order repaired and checked by reading, not compiled, not run on hardware
//
#include <hip/hip_runtime.h>

// ---------------------------------------------------------------------------
// BiLSTM-CRF forward loss for MI455X (gfx1250, wave32, WMMA).
//   T0=128 (time), T1=512 (LSTM batch), E=256, H=256, NT=9
//   CRF views logits as [B=128 sequences, L=512 positions, 9 tags].
// Strategy:
//   1) Hoist input projection out of the scan: one big f16 WMMA GEMM per
//      direction, pre_gates[t,b,:] = x . W_ih^T + b   (65536x256 * 256x1024)
//   2) 128 sequential recurrence steps per direction, each a WMMA GEMM
//      G = pre[t] + h_prev . W_hh^T  followed by fused LSTM gate math.
//   3) Tiny VALU kernels for the 9-tag output projection and CRF forward
//      algorithm (single block, fused reduction -> scalar NLL).
// GEMM: each wave owns a 32x64 C tile (2x4 WMMA subtiles, 8 f32 accumulators)
// and is software-pipelined: fragment loads for k+32 are issued before the 8
// WMMAs of step k, so VMEM overlaps matrix math (loads return in order).
// Weights are f16, pre-packed into the native B-fragment lane layout so the
// inner loop does only contiguous 128-bit loads.
// ---------------------------------------------------------------------------

typedef __attribute__((ext_vector_type(16))) _Float16 v16h;
typedef __attribute__((ext_vector_type(8)))  _Float16 v8h;
typedef __attribute__((ext_vector_type(8)))  float    v8f;

constexpr int kT0 = 128, kT1 = 512, kE = 256, kH = 256, kNT = 9;
constexpr int kG  = 4 * kH;          // 1024 (gate width)
constexpr int kM  = kT0 * kT1;       // 65536 rows in the hoisted GEMM

__device__ __forceinline__ v16h cat8(v8h lo, v8h hi) {
  return __builtin_shufflevector(lo, hi, 0,1,2,3,4,5,6,7,8,9,10,11,12,13,14,15);
}

// A fragment: 16x32 f16 tile at (m0,k0) of row-major A[* x lda].
// ISA layout: lane l<16 holds row m0+l, K = k0+{0..7} and k0+{16..23};
// lanes 16..31 hold K = k0+{8..15} and k0+{24..31}. Two 16B contiguous loads.
__device__ __forceinline__ v16h load_a_frag(const _Float16* A, int lda, int m0, int k0) {
  const int lane = threadIdx.x & 31;
  const int m  = m0 + (lane & 15);
  const int kb = k0 + ((lane >> 4) << 3);
  const v8h* p0 = reinterpret_cast<const v8h*>(A + (size_t)m * lda + kb);
  const v8h* p1 = reinterpret_cast<const v8h*>(A + (size_t)m * lda + kb + 16);
  return cat8(*p0, *p1);
}

// B fragment from the pre-packed layout: halves stored so each lane reads
// 16 contiguous f16 (32B). Packed index = (((kt*ntn + nt)*32 + lane)*16 + i).
__device__ __forceinline__ v16h load_b_frag(const _Float16* Bp, int ntn, int kt, int nt) {
  const int lane = threadIdx.x & 31;
  const v16h* p = reinterpret_cast<const v16h*>(
      Bp + ((((size_t)kt * ntn + nt) * 32 + lane) << 4));
  return *p;
}

// ---------------- small utility kernels ----------------

__global__ void cvt_f32_to_f16_kernel(const float* __restrict__ in,
                                      _Float16* __restrict__ out, int n) {
  int i = blockIdx.x * blockDim.x + threadIdx.x;
  if (i < n) out[i] = (_Float16)in[i];
}

__global__ void zero_u32_kernel(unsigned int* __restrict__ p, int n) {
  int i = blockIdx.x * blockDim.x + threadIdx.x;
  if (i < n) p[i] = 0u;
}

// Pack W [N x K] row-major f32 (as given: W(n,k) = W[n*K+k], i.e. we need
// B = W^T) into B-fragment-native order, converting to f16.
__global__ void pack_w_kernel(const float* __restrict__ W,
                              _Float16* __restrict__ Bp, int N, int K) {
  int idx = blockIdx.x * blockDim.x + threadIdx.x;   // one thread per half
  if (idx >= N * K) return;
  const int i    = idx & 15;
  const int lane = (idx >> 4) & 31;
  const int tile = idx >> 9;                         // kt*ntn + nt
  const int ntn  = N >> 4;
  const int kt = tile / ntn, nt = tile % ntn;
  const int n = nt * 16 + (lane & 15);
  const int k = kt * 32 + ((lane >> 4) << 3) + (i & 7) + ((i >> 3) << 4);
  Bp[idx] = (_Float16)W[(size_t)n * K + k];
}

// ---------------- WMMA GEMM: C[M,N] = A[M,K] * Bpack (+bias)(+addend) ------
// One wave per 32x64 C tile (2 M-subtiles x 4 N-subtiles), 8 waves / block.
// Software-pipelined: next k-step fragments are loaded before current WMMAs.
__global__ void gemm_wmma_kernel(const _Float16* __restrict__ A,
                                 const _Float16* __restrict__ Bp,
                                 const _Float16* __restrict__ addend,
                                 const float*    __restrict__ bias,
                                 float*    __restrict__ Cf32,
                                 _Float16* __restrict__ Cf16,
                                 int M, int N, int K) {
  const int wave  = blockIdx.x * (blockDim.x >> 5) + (threadIdx.x >> 5);
  const int ntn16 = N >> 4;                   // 16-wide tiles along N
  const int ntn64 = N >> 6;                   // 64-wide macro tiles along N
  const int total = (M >> 5) * ntn64;
  if (wave >= total) return;                  // wave-uniform: EXEC stays all-1s
  const int m0 = (wave / ntn64) << 5;
  const int n0 = (wave % ntn64) << 6;
  const int nt0 = n0 >> 4;
  const int lane = threadIdx.x & 31;

  v8f acc[2][4];
#pragma unroll
  for (int mi = 0; mi < 2; ++mi)
#pragma unroll
    for (int ni = 0; ni < 4; ++ni) acc[mi][ni] = (v8f){};

  // prologue: fragments for k0 = 0
  v16h a0 = load_a_frag(A, K, m0,      0);
  v16h a1 = load_a_frag(A, K, m0 + 16, 0);
  v16h b0 = load_b_frag(Bp, ntn16, 0, nt0 + 0);
  v16h b1 = load_b_frag(Bp, ntn16, 0, nt0 + 1);
  v16h b2 = load_b_frag(Bp, ntn16, 0, nt0 + 2);
  v16h b3 = load_b_frag(Bp, ntn16, 0, nt0 + 3);

  for (int k0 = 0; k0 + 32 < K; k0 += 32) {
    const int kn = k0 + 32;
    // issue next-step loads first so they overlap the 8 WMMAs below
    v16h na0 = load_a_frag(A, K, m0,      kn);
    v16h na1 = load_a_frag(A, K, m0 + 16, kn);
    v16h nb0 = load_b_frag(Bp, ntn16, kn >> 5, nt0 + 0);
    v16h nb1 = load_b_frag(Bp, ntn16, kn >> 5, nt0 + 1);
    v16h nb2 = load_b_frag(Bp, ntn16, kn >> 5, nt0 + 2);
    v16h nb3 = load_b_frag(Bp, ntn16, kn >> 5, nt0 + 3);

    acc[0][0] = __builtin_amdgcn_wmma_f32_16x16x32_f16(false, a0, false, b0, (short)0, acc[0][0], false, false);
    acc[0][1] = __builtin_amdgcn_wmma_f32_16x16x32_f16(false, a0, false, b1, (short)0, acc[0][1], false, false);
    acc[0][2] = __builtin_amdgcn_wmma_f32_16x16x32_f16(false, a0, false, b2, (short)0, acc[0][2], false, false);
    acc[0][3] = __builtin_amdgcn_wmma_f32_16x16x32_f16(false, a0, false, b3, (short)0, acc[0][3], false, false);
    acc[1][0] = __builtin_amdgcn_wmma_f32_16x16x32_f16(false, a1, false, b0, (short)0, acc[1][0], false, false);
    acc[1][1] = __builtin_amdgcn_wmma_f32_16x16x32_f16(false, a1, false, b1, (short)0, acc[1][1], false, false);
    acc[1][2] = __builtin_amdgcn_wmma_f32_16x16x32_f16(false, a1, false, b2, (short)0, acc[1][2], false, false);
    acc[1][3] = __builtin_amdgcn_wmma_f32_16x16x32_f16(false, a1, false, b3, (short)0, acc[1][3], false, false);

    a0 = na0; a1 = na1; b0 = nb0; b1 = nb1; b2 = nb2; b3 = nb3;
  }
  // epilogue k-step
  acc[0][0] = __builtin_amdgcn_wmma_f32_16x16x32_f16(false, a0, false, b0, (short)0, acc[0][0], false, false);
  acc[0][1] = __builtin_amdgcn_wmma_f32_16x16x32_f16(false, a0, false, b1, (short)0, acc[0][1], false, false);
  acc[0][2] = __builtin_amdgcn_wmma_f32_16x16x32_f16(false, a0, false, b2, (short)0, acc[0][2], false, false);
  acc[0][3] = __builtin_amdgcn_wmma_f32_16x16x32_f16(false, a0, false, b3, (short)0, acc[0][3], false, false);
  acc[1][0] = __builtin_amdgcn_wmma_f32_16x16x32_f16(false, a1, false, b0, (short)0, acc[1][0], false, false);
  acc[1][1] = __builtin_amdgcn_wmma_f32_16x16x32_f16(false, a1, false, b1, (short)0, acc[1][1], false, false);
  acc[1][2] = __builtin_amdgcn_wmma_f32_16x16x32_f16(false, a1, false, b2, (short)0, acc[1][2], false, false);
  acc[1][3] = __builtin_amdgcn_wmma_f32_16x16x32_f16(false, a1, false, b3, (short)0, acc[1][3], false, false);

  // C layout per subtile: VGPR r -> M = base + r + 8*(lane>=16), N = n + (lane&15)
  const int nlo   = lane & 15;
  const int mhalf = (lane >> 4) << 3;
#pragma unroll
  for (int mi = 0; mi < 2; ++mi) {
    const int mbase = m0 + mi * 16 + mhalf;
#pragma unroll
    for (int ni = 0; ni < 4; ++ni) {
      const int n = n0 + ni * 16 + nlo;
      const float bn = bias ? bias[n] : 0.0f;
#pragma unroll
      for (int r = 0; r < 8; ++r) {
        const size_t off = (size_t)(mbase + r) * N + n;
        float v = acc[mi][ni][r] + bn;
        if (addend) v += (float)addend[off];
        if (Cf32) Cf32[off] = v;
        if (Cf16) Cf16[off] = (_Float16)v;
      }
    }
  }
}

// ---------------- fused LSTM gate math for one time step -------------------
__device__ __forceinline__ float sigf(float x) { return 1.0f / (1.0f + __expf(-x)); }

__global__ void lstm_gate_kernel(const float* __restrict__ G,   // [T1, 4H]
                                 float*    __restrict__ c,      // [T1, H]
                                 _Float16* __restrict__ hbuf,   // [T1, H] f16 (next A)
                                 float*    __restrict__ hs) {   // [T1, H] slot t
  int idx = blockIdx.x * blockDim.x + threadIdx.x;
  if (idx >= kT1 * kH) return;
  const int b = idx / kH, h = idx - b * kH;
  const float* g = G + (size_t)b * kG;
  const float i_g = sigf(g[h]);
  const float f_g = sigf(g[kH + h]);
  const float g_g = tanhf(g[2 * kH + h]);
  const float o_g = sigf(g[3 * kH + h]);
  const float cn  = f_g * c[idx] + i_g * g_g;
  c[idx] = cn;
  const float hn = o_g * tanhf(cn);
  hbuf[idx] = (_Float16)hn;
  hs[idx]   = hn;
}

// ---------------- output projection: logits = [hf|hb] . W_out^T + b --------
__global__ void logits_kernel(const float* __restrict__ hf,
                              const float* __restrict__ hb,
                              const float* __restrict__ Wout,   // [9, 512]
                              const float* __restrict__ bout,   // [9]
                              float* __restrict__ logits) {     // [65536, 9]
  __shared__ float Wl[kNT * 2 * kH];                            // 18 KB
  for (int i = threadIdx.x; i < kNT * 2 * kH; i += blockDim.x) Wl[i] = Wout[i];
  __syncthreads();
  int idx = blockIdx.x * blockDim.x + threadIdx.x;
  if (idx >= kM) return;
  const float* xf = hf + (size_t)idx * kH;
  const float* xb = hb + (size_t)idx * kH;
  float acc[kNT];
#pragma unroll
  for (int j = 0; j < kNT; ++j) acc[j] = bout[j];
  for (int k = 0; k < kH; ++k) {
    const float x = xf[k];
#pragma unroll
    for (int j = 0; j < kNT; ++j) acc[j] += x * Wl[j * 2 * kH + k];
  }
  for (int k = 0; k < kH; ++k) {
    const float x = xb[k];
#pragma unroll
    for (int j = 0; j < kNT; ++j) acc[j] += x * Wl[j * 2 * kH + kH + k];
  }
#pragma unroll
  for (int j = 0; j < kNT; ++j) logits[(size_t)idx * kNT + j] = acc[j];
}

// ---------------- CRF NLL: 128 sequences of length 512, 9 tags -------------
// One block of 128 threads (one sequence per thread), fused mean-reduction.
__global__ void crf_kernel(const float* __restrict__ logits,   // [128,512,9]
                           const int* __restrict__ labels,     // [128,512]
                           const unsigned char* __restrict__ mask,
                           const float* __restrict__ start,
                           const float* __restrict__ endt,
                           const float* __restrict__ trans,    // [9,9]
                           float* __restrict__ out) {
  __shared__ float tr[kNT * kNT];
  __shared__ float red[kT0];
  const int b = threadIdx.x;           // 0..127
  if (b < kNT * kNT) tr[b] = trans[b];
  __syncthreads();

  const float* lg = logits + (size_t)b * kT1 * kNT;
  const int*   lb = labels + (size_t)b * kT1;
  const unsigned char* mk = mask + (size_t)b * kT1;

  float alpha[kNT];
#pragma unroll
  for (int j = 0; j < kNT; ++j) alpha[j] = start[j] + lg[j];

  float numer = start[lb[0]] + (mk[0] ? lg[lb[0]] : 0.0f);
  int prev = lb[0];
  int seqlen = mk[0] ? 1 : 0;

  for (int l = 1; l < kT1; ++l) {
    const float* e = lg + (size_t)l * kNT;
    const int lab = lb[l];
    const bool mt = mk[l] != 0;
    if (mt) { numer += tr[prev * kNT + lab] + e[lab]; seqlen++; }
    prev = lab;
    float nxt[kNT];
#pragma unroll
    for (int j = 0; j < kNT; ++j) {
      float mx = -1e30f;
#pragma unroll
      for (int i = 0; i < kNT; ++i) mx = fmaxf(mx, alpha[i] + tr[i * kNT + j]);
      float s = 0.0f;
#pragma unroll
      for (int i = 0; i < kNT; ++i) s += __expf(alpha[i] + tr[i * kNT + j] - mx);
      nxt[j] = mx + __logf(s) + e[j];
    }
    if (mt) {
#pragma unroll
      for (int j = 0; j < kNT; ++j) alpha[j] = nxt[j];
    }
  }
  numer += endt[lb[seqlen - 1]];
  float mx = -1e30f;
#pragma unroll
  for (int j = 0; j < kNT; ++j) mx = fmaxf(mx, alpha[j] + endt[j]);
  float s = 0.0f;
#pragma unroll
  for (int j = 0; j < kNT; ++j) s += __expf(alpha[j] + endt[j] - mx);
  const float denom = mx + __logf(s);

  red[b] = numer - denom;
  __syncthreads();
  for (int st = kT0 / 2; st > 0; st >>= 1) {
    if (b < st) red[b] += red[b + st];
    __syncthreads();
  }
  if (b == 0) out[0] = -red[0] / (float)kT0;
}

// ---------------------------------------------------------------------------

static inline size_t alignUp(size_t x) { return (x + 255) & ~(size_t)255; }

extern "C" void kernel_launch(void* const* d_in, const int* in_sizes, int n_in,
                              void* d_out, int out_size, void* d_ws, size_t ws_size,
                              hipStream_t stream) {
  (void)in_sizes; (void)n_in; (void)out_size; (void)ws_size;
  const float* emb    = (const float*)d_in[0];
  const int*   labels = (const int*)d_in[1];
  const unsigned char* mask = (const unsigned char*)d_in[2];
  const float* w_ih_f = (const float*)d_in[3];
  const float* w_hh_f = (const float*)d_in[4];
  const float* b_f    = (const float*)d_in[5];
  const float* w_ih_b = (const float*)d_in[6];
  const float* w_hh_b = (const float*)d_in[7];
  const float* b_b    = (const float*)d_in[8];
  const float* W_out  = (const float*)d_in[9];
  const float* b_out  = (const float*)d_in[10];
  const float* startt = (const float*)d_in[11];
  const float* endt   = (const float*)d_in[12];
  const float* transs = (const float*)d_in[13];
  float* out = (float*)d_out;

  // -------- workspace carve-up --------
  char* ws = (char*)d_ws;
  size_t off = 0;
  auto take = [&](size_t bytes) { char* p = ws + off; off = alignUp(off + bytes); return p; };
  _Float16* Xh    = (_Float16*)take((size_t)kM * kE * 2);        // emb f16
  _Float16* BpIhF = (_Float16*)take((size_t)kG * kE * 2);
  _Float16* BpIhB = (_Float16*)take((size_t)kG * kE * 2);
  _Float16* BpHhF = (_Float16*)take((size_t)kG * kH * 2);
  _Float16* BpHhB = (_Float16*)take((size_t)kG * kH * 2);
  _Float16* preF  = (_Float16*)take((size_t)kM * kG * 2);        // 128 MB
  _Float16* preB  = (_Float16*)take((size_t)kM * kG * 2);        // 128 MB
  float*    hsF   = (float*)take((size_t)kM * kH * 4);           // 64 MB
  float*    hsB   = (float*)take((size_t)kM * kH * 4);           // 64 MB
  _Float16* hbF   = (_Float16*)take((size_t)kT1 * kH * 2);
  _Float16* hbB   = (_Float16*)take((size_t)kT1 * kH * 2);
  float*    cF    = (float*)take((size_t)kT1 * kH * 4);
  float*    cB    = (float*)take((size_t)kT1 * kH * 4);
  float*    GF    = (float*)take((size_t)kT1 * kG * 4);
  float*    GB    = (float*)take((size_t)kT1 * kG * 4);
  float*    lgts  = (float*)take((size_t)kM * kNT * 4);

  // -------- 1) convert embeddings to f16, pack weights into B layout ------
  {
    int n = kM * kE;
    cvt_f32_to_f16_kernel<<<(n + 255) / 256, 256, 0, stream>>>(emb, Xh, n);
  }
  {
    int n = kG * kE;   // 262144 halves each
    pack_w_kernel<<<(n + 255) / 256, 256, 0, stream>>>(w_ih_f, BpIhF, kG, kE);
    pack_w_kernel<<<(n + 255) / 256, 256, 0, stream>>>(w_ih_b, BpIhB, kG, kE);
    pack_w_kernel<<<(n + 255) / 256, 256, 0, stream>>>(w_hh_f, BpHhF, kG, kH);
    pack_w_kernel<<<(n + 255) / 256, 256, 0, stream>>>(w_hh_b, BpHhB, kG, kH);
  }
  // -------- 2) zero initial h (f16) and c (f32) ---------------------------
  {
    int nh = kT1 * kH / 2;   // f16 pairs as u32
    int nc = kT1 * kH;
    zero_u32_kernel<<<(nh + 255) / 256, 256, 0, stream>>>((unsigned int*)hbF, nh);
    zero_u32_kernel<<<(nh + 255) / 256, 256, 0, stream>>>((unsigned int*)hbB, nh);
    zero_u32_kernel<<<(nc + 255) / 256, 256, 0, stream>>>((unsigned int*)cF, nc);
    zero_u32_kernel<<<(nc + 255) / 256, 256, 0, stream>>>((unsigned int*)cB, nc);
  }
  // -------- 3) hoisted input projection: pre = X . W_ih^T + b (WMMA) ------
  {
    int waves  = (kM / 32) * (kG / 64);         // 32768 waves (32x64 tiles)
    int blocks = waves / 8;                     // 4096 blocks of 256 threads
    gemm_wmma_kernel<<<blocks, 256, 0, stream>>>(Xh, BpIhF, nullptr, b_f,
                                                 nullptr, preF, kM, kG, kE);
    gemm_wmma_kernel<<<blocks, 256, 0, stream>>>(Xh, BpIhB, nullptr, b_b,
                                                 nullptr, preB, kM, kG, kE);
  }
  // -------- 4) 128 recurrence steps per direction (WMMA + fused gates) ----
  {
    int waves  = (kT1 / 32) * (kG / 64);        // 256 waves per step
    int blocks = waves / 8;                     // 32 blocks
    int gateN  = kT1 * kH;
    for (int step = 0; step < kT0; ++step) {
      const int tf = step;            // forward time index
      const int tb = kT0 - 1 - step;  // backward time index
      gemm_wmma_kernel<<<blocks, 256, 0, stream>>>(
          hbF, BpHhF, preF + (size_t)tf * kT1 * kG, nullptr,
          GF, nullptr, kT1, kG, kH);
      lstm_gate_kernel<<<(gateN + 255) / 256, 256, 0, stream>>>(
          GF, cF, hbF, hsF + (size_t)tf * kT1 * kH);
      gemm_wmma_kernel<<<blocks, 256, 0, stream>>>(
          hbB, BpHhB, preB + (size_t)tb * kT1 * kG, nullptr,
          GB, nullptr, kT1, kG, kH);
      lstm_gate_kernel<<<(gateN + 255) / 256, 256, 0, stream>>>(
          GB, cB, hbB, hsB + (size_t)tb * kT1 * kH);
    }
  }
  // -------- 5) output projection + CRF --------------------------------
  logits_kernel<<<(kM + 255) / 256, 256, 0, stream>>>(hsF, hsB, W_out, b_out, lgts);
  crf_kernel<<<1, kT0, 0, stream>>>(lgts, labels, mask, startt, endt, transs, out);
}
